// MultiHeadAttention_28312424415617
// MI455X (gfx1250) — matrix operand order repaired and checked by reading, not compiled, not run on hardware
//
#include <hip/hip_runtime.h>

// ---------------------------------------------------------------------------
// MultiHeadAttention forward for MI455X (gfx1250, wave32, WMMA f16 16x16x32)
//   d = 1024, heads = 16, head_dim = 64, batch = 2, seq = 2048
// Pipeline (all on `stream`), preferred path (ws >= 42 MB):
//   0) cvt_f16    : activations / weights f32 -> f16 staging (once each)
//   1) proj_gemm  : Q16 = f16(pre_q @ Wq^T + bq)            [4096 x 1024] f16
//   2) proj_gemm  : K16 = f16(pre_k @ Wk^T + bk)            [4096 x 1024] f16
//   3) proj_gemm  : Vt16 = f16(pre_v @ Wv^T + bv) TRANSPOSED to [b][h][64][2048]
//   4) attn_kernel: flash attention (online softmax), A16 f16 [4096 x 1024]
//   5) proj_gemm  : out = f32(A16 @ Wo^T + bo)              [4096 x 1024] f32
// The pre-conversion keeps the GEMM K-loop free of v_cvt VALU traffic so the
// v_wmma (XDL/TRANS) stream can co-execute with address math, and halves the
// bytes re-fetched per K-sweep. Fallback path (ws >= 32 MB) converts inline.
// ---------------------------------------------------------------------------

#define D_MODEL   1024
#define NUM_HEADS 16
#define HEAD_DIM  64
#define BATCH     2
#define SEQ       2048
#define ROWS      (BATCH * SEQ)   // 4096

typedef __attribute__((ext_vector_type(16))) _Float16 v16h;
typedef __attribute__((ext_vector_type(8)))  _Float16 v8h;
typedef __attribute__((ext_vector_type(8)))  float    v8f;

static __device__ __forceinline__ v8f wmma16x16x32(v16h a, v16h b, v8f c) {
  // D = A(16x32 f16) * B(32x16 f16) + C(16x16 f32)
  return __builtin_amdgcn_wmma_f32_16x16x32_f16(
      /*neg_a=*/false, a, /*neg_b=*/false, b,
      /*c_mod=*/(short)0, c, /*reuse_a=*/false, /*reuse_b=*/false);
}

// Convert 8 contiguous f32 into v16h positions [base .. base+7]
static __device__ __forceinline__ void cvt_f32x8(v16h& d, int base, const float* p) {
  float4 f0 = *(const float4*)p;
  float4 f1 = *(const float4*)(p + 4);
  d[base + 0] = (_Float16)f0.x; d[base + 1] = (_Float16)f0.y;
  d[base + 2] = (_Float16)f0.z; d[base + 3] = (_Float16)f0.w;
  d[base + 4] = (_Float16)f1.x; d[base + 5] = (_Float16)f1.y;
  d[base + 6] = (_Float16)f1.z; d[base + 7] = (_Float16)f1.w;
}

static __device__ __forceinline__ v16h cat8(v8h a, v8h b) {
  return __builtin_shufflevector(a, b, 0,1,2,3,4,5,6,7,8,9,10,11,12,13,14,15);
}

// ---------------------------------------------------------------------------
// Bulk f32 -> f16 conversion: 8 elements/thread, coalesced b128 in / b128 out.
// ---------------------------------------------------------------------------
__global__ __launch_bounds__(256) void cvt_f16_kernel(
    const float* __restrict__ src, _Float16* __restrict__ dst)
{
  const size_t i = ((size_t)blockIdx.x * 256 + threadIdx.x) * 8;
  float4 f0 = *(const float4*)(src + i);
  float4 f1 = *(const float4*)(src + i + 4);
  v8h h;
  h[0] = (_Float16)f0.x; h[1] = (_Float16)f0.y;
  h[2] = (_Float16)f0.z; h[3] = (_Float16)f0.w;
  h[4] = (_Float16)f1.x; h[5] = (_Float16)f1.y;
  h[6] = (_Float16)f1.z; h[7] = (_Float16)f1.w;
  *(v8h*)(dst + i) = h;
}

// ---------------------------------------------------------------------------
// Y = X @ W^T + bias.  X: [4096, 1024], W: [1024,1024] row-major (f16 or f32).
// Block = 128 threads (4 waves); block tile = 64(M) x 128(N); wave = 16 x 128.
// K loop in steps of 32 -> 8 v_wmma_f32_16x16x32_f16 per iteration per wave,
// with a single A-tile load (2x16B) amortized over all 8 N-chunks.
// OUT_VT: scatter output as f16 [b][h][HEAD_DIM][SEQ] (for attention V operand).
// ---------------------------------------------------------------------------
#define GEMM_NC 8   // N chunks of 16 per wave tile

template <bool A_F16, bool W_F16, bool OUT_F32, bool OUT_VT>
__global__ __launch_bounds__(128) void proj_gemm(
    const void* __restrict__ Xv, const void* __restrict__ Wv,
    const float* __restrict__ bias, void* __restrict__ Yv)
{
  const int lane = threadIdx.x & 31;
  const int wave = threadIdx.x >> 5;
  const int hi   = lane >> 4;      // lane half (0: lanes 0-15, 1: lanes 16-31)
  const int lo   = lane & 15;
  const int m0   = blockIdx.x * 64 + wave * 16;
  const int n0   = blockIdx.y * (16 * GEMM_NC);
  const int arow = m0 + lo;

  v8f acc[GEMM_NC] = {};

  for (int kb = 0; kb < D_MODEL; kb += 32) {
    // ---- A operand: 16x32, lane holds K runs {kb+hi*8 .. +7} and {+16 .. +23}
    v16h a;
    if (A_F16) {
      const _Float16* p = (const _Float16*)Xv + (size_t)arow * D_MODEL + kb + hi * 8;
      a = cat8(*(const v8h*)p, *(const v8h*)(p + 16));
    } else {
      const float* p = (const float*)Xv + (size_t)arow * D_MODEL + kb + hi * 8;
      cvt_f32x8(a, 0, p);
      cvt_f32x8(a, 8, p + 16);
    }
    // ---- B operand per 16-wide N chunk: lane lo = W row (=output col),
    //      K runs {kb+hi*8 .. +7} and {+16 .. +23} (same order as A).
#pragma unroll
    for (int nc = 0; nc < GEMM_NC; ++nc) {
      v16h b;
      if (W_F16) {
        const _Float16* wp =
            (const _Float16*)Wv + (size_t)(n0 + nc * 16 + lo) * D_MODEL + kb + hi * 8;
        b = cat8(*(const v8h*)wp, *(const v8h*)(wp + 16));
      } else {
        const float* wp =
            (const float*)Wv + (size_t)(n0 + nc * 16 + lo) * D_MODEL + kb + hi * 8;
        cvt_f32x8(b, 0, wp);
        cvt_f32x8(b, 8, wp + 16);
      }
      acc[nc] = wmma16x16x32(a, b, acc[nc]);
    }
  }

  // ---- epilogue: C layout -> element (m = r + 8*hi, n = lo) in vgpr r
#pragma unroll
  for (int nc = 0; nc < GEMM_NC; ++nc) {
    const int n  = n0 + nc * 16 + lo;
    const float bn = bias[n];
#pragma unroll
    for (int r = 0; r < 8; ++r) {
      const int m = m0 + r + 8 * hi;
      const float val = acc[nc][r] + bn;
      if (OUT_F32) {
        ((float*)Yv)[(size_t)m * D_MODEL + n] = val;
      } else if (OUT_VT) {
        const int bb = m >> 11;      // batch
        const int ss = m & (SEQ - 1);
        const int hh = n >> 6;       // head
        const int dd = n & (HEAD_DIM - 1);
        ((_Float16*)Yv)[(((size_t)(bb * NUM_HEADS + hh)) * HEAD_DIM + dd) * SEQ + ss] =
            (_Float16)val;
      } else {
        ((_Float16*)Yv)[(size_t)m * D_MODEL + n] = (_Float16)val;
      }
    }
  }
}

// ---------------------------------------------------------------------------
// Flash attention. Grid: (SEQ/64, BATCH*NUM_HEADS), block 128 (4 waves).
// Each wave owns 16 query rows; key loop in chunks of 32.
// Q/K: f16 [4096,1024] row-major; Vt: f16 [b][h][64][2048]; A: f16 [4096,1024].
// ---------------------------------------------------------------------------
__global__ __launch_bounds__(128) void attn_kernel(
    const _Float16* __restrict__ Q, const _Float16* __restrict__ K,
    const _Float16* __restrict__ Vt, const unsigned char* __restrict__ mask,
    _Float16* __restrict__ A)
{
  __shared__ __align__(64) _Float16 plds[4][16 * 32];  // per-wave P staging

  const int lane = threadIdx.x & 31;
  const int wave = threadIdx.x >> 5;
  const int hi = lane >> 4, lo = lane & 15;
  const int bh = blockIdx.y;
  const int b = bh >> 4, h = bh & 15;
  const int m0 = blockIdx.x * 64 + wave * 16;          // query row base (within SEQ)
  const size_t batchBase = (size_t)b * SEQ * D_MODEL;  // f16 element offset
  const int colQ = h * HEAD_DIM;
  const float scale = 0.125f;                          // 1/sqrt(64)

  // ---- Q A-operand: two 16x32 tiles over head_dim (held for the whole loop)
  v16h aq[2];
  {
    const _Float16* qp = Q + batchBase + (size_t)(m0 + lo) * D_MODEL + colQ;
#pragma unroll
    for (int c = 0; c < 2; ++c) {
      const _Float16* p = qp + c * 32 + hi * 8;
      aq[c] = cat8(*(const v8h*)p, *(const v8h*)(p + 16));
    }
  }

  float mrow[8], lrow[8];
  v8f o[4] = {};
#pragma unroll
  for (int r = 0; r < 8; ++r) { mrow[r] = -__builtin_inff(); lrow[r] = 0.f; }

  for (int kk = 0; kk < SEQ; kk += 32) {
    // ---- S = Q K^T for 32 keys: two 16x16 tiles, contraction over head_dim
    v8f s[2] = {};
#pragma unroll
    for (int nh = 0; nh < 2; ++nh) {
      const _Float16* kp = K + batchBase + (size_t)(kk + nh * 16 + lo) * D_MODEL + colQ;
#pragma unroll
      for (int c = 0; c < 2; ++c) {
        v16h bk = cat8(*(const v8h*)(kp + c * 32 + hi * 8), *(const v8h*)(kp + c * 32 + hi * 8 + 16));
        s[nh] = wmma16x16x32(aq[c], bk, s[nh]);
      }
    }
    // ---- scale + mask
#pragma unroll
    for (int nh = 0; nh < 2; ++nh) {
#pragma unroll
      for (int r = 0; r < 8; ++r) {
        const int m = m0 + r + 8 * hi;
        const int n = kk + nh * 16 + lo;
        float v = s[nh][r] * scale;
        if (!mask[(size_t)m * SEQ + n]) v = -__builtin_inff();
        s[nh][r] = v;
      }
    }
    // ---- online softmax (row r lives in lanes of one 16-lane half)
#pragma unroll
    for (int r = 0; r < 8; ++r) {
      float t = fmaxf(s[0][r], s[1][r]);
#pragma unroll
      for (int d = 1; d < 16; d <<= 1) t = fmaxf(t, __shfl_xor(t, d, 32));
      const float mnew  = fmaxf(mrow[r], t);
      const float alpha = __expf(mrow[r] - mnew);
      mrow[r] = mnew;
      const float p0 = __expf(s[0][r] - mnew);
      const float p1 = __expf(s[1][r] - mnew);
      s[0][r] = p0; s[1][r] = p1;
      float rs = p0 + p1;
#pragma unroll
      for (int d = 1; d < 16; d <<= 1) rs += __shfl_xor(rs, d, 32);
      lrow[r] = lrow[r] * alpha + rs;
#pragma unroll
      for (int dc = 0; dc < 4; ++dc) o[dc][r] *= alpha;
    }
    // ---- P (f32 C-layout) -> LDS row-major f16 16x32
#pragma unroll
    for (int r = 0; r < 8; ++r) {
      const int ml = r + 8 * hi;
      plds[wave][ml * 32 + lo]      = (_Float16)s[0][r];
      plds[wave][ml * 32 + 16 + lo] = (_Float16)s[1][r];
    }
    // ---- reload P in A-operand layout (same-wave LDS is in-order)
    v16h pa;
    {
      const _Float16* p = &plds[wave][lo * 32 + hi * 8];
      pa = cat8(*(const v8h*)p, *(const v8h*)(p + 16));
    }
    // ---- O += P V : Vt gives contiguous keys per lane (B operand)
#pragma unroll
    for (int dc = 0; dc < 4; ++dc) {
      const _Float16* vp =
          Vt + ((size_t)(b * NUM_HEADS + h) * HEAD_DIM + dc * 16 + lo) * SEQ + kk + hi * 8;
      v16h bv = cat8(*(const v8h*)vp, *(const v8h*)(vp + 16));
      o[dc] = wmma16x16x32(pa, bv, o[dc]);
    }
  }

  // ---- normalize and store attention output (f16, head-interleaved rows)
#pragma unroll
  for (int r = 0; r < 8; ++r) {
    const float inv = 1.0f / lrow[r];
    const int m = m0 + r + 8 * hi;
#pragma unroll
    for (int dc = 0; dc < 4; ++dc) {
      A[batchBase + (size_t)m * D_MODEL + colQ + dc * 16 + lo] =
          (_Float16)(o[dc][r] * inv);
    }
  }
}

// ---------------------------------------------------------------------------
extern "C" void kernel_launch(void* const* d_in, const int* in_sizes, int n_in,
                              void* d_out, int out_size, void* d_ws, size_t ws_size,
                              hipStream_t stream)
{
  const float*         pre_q = (const float*)d_in[0];
  const float*         pre_k = (const float*)d_in[1];
  const float*         pre_v = (const float*)d_in[2];
  const unsigned char* mask  = (const unsigned char*)d_in[3];
  const float* Wq = (const float*)d_in[4];
  const float* bq = (const float*)d_in[5];
  const float* Wk = (const float*)d_in[6];
  const float* bk = (const float*)d_in[7];
  const float* Wv = (const float*)d_in[8];
  const float* bv = (const float*)d_in[9];
  const float* Wo = (const float*)d_in[10];
  const float* bo = (const float*)d_in[11];
  (void)in_sizes; (void)n_in; (void)out_size;

  const size_t actElems = (size_t)ROWS * D_MODEL;       // 4M
  const size_t wElems   = (size_t)D_MODEL * D_MODEL;    // 1M
  const size_t bufBytes = actElems * sizeof(_Float16);  // 8 MB

  char* ws = (char*)d_ws;
  _Float16* Q16  = (_Float16*)(ws + 0 * bufBytes);
  _Float16* K16  = (_Float16*)(ws + 1 * bufBytes);
  _Float16* Vt16 = (_Float16*)(ws + 2 * bufBytes);  // [b][h][64][2048]
  _Float16* A16  = (_Float16*)(ws + 3 * bufBytes);
  _Float16* X16  = (_Float16*)(ws + 4 * bufBytes);               // reused staging
  _Float16* W16  = (_Float16*)(ws + 5 * bufBytes);               // reused staging
  const size_t fullBytes = 5 * bufBytes + wElems * sizeof(_Float16);  // 42 MB

  dim3 gproj(ROWS / 64, D_MODEL / (16 * GEMM_NC));
  dim3 gattn(SEQ / 64, BATCH * NUM_HEADS);
  const int cvtActBlocks = (int)(actElems / (256 * 8));
  const int cvtWBlocks   = (int)(wElems   / (256 * 8));

  if (ws_size >= fullBytes) {
    // Preferred: pre-convert operands once; GEMM K-loop is pure load+WMMA.
    cvt_f16_kernel<<<cvtActBlocks, 256, 0, stream>>>(pre_q, X16);
    cvt_f16_kernel<<<cvtWBlocks,   256, 0, stream>>>(Wq, W16);
    proj_gemm<true, true, false, false><<<gproj, 128, 0, stream>>>(X16, W16, bq, Q16);

    cvt_f16_kernel<<<cvtActBlocks, 256, 0, stream>>>(pre_k, X16);
    cvt_f16_kernel<<<cvtWBlocks,   256, 0, stream>>>(Wk, W16);
    proj_gemm<true, true, false, false><<<gproj, 128, 0, stream>>>(X16, W16, bk, K16);

    cvt_f16_kernel<<<cvtActBlocks, 256, 0, stream>>>(pre_v, X16);
    cvt_f16_kernel<<<cvtWBlocks,   256, 0, stream>>>(Wv, W16);
    proj_gemm<true, true, false, true ><<<gproj, 128, 0, stream>>>(X16, W16, bv, Vt16);

    attn_kernel<<<gattn, 128, 0, stream>>>(Q16, K16, Vt16, mask, A16);

    cvt_f16_kernel<<<cvtWBlocks, 256, 0, stream>>>(Wo, W16);
    proj_gemm<true, true, true, false><<<gproj, 128, 0, stream>>>(A16, W16, bo, d_out);
  } else {
    // Fallback (32 MB ws): convert inline inside the GEMM.
    proj_gemm<false, false, false, false><<<gproj, 128, 0, stream>>>(pre_q, Wq, bq, Q16);
    proj_gemm<false, false, false, false><<<gproj, 128, 0, stream>>>(pre_k, Wk, bk, K16);
    proj_gemm<false, false, false, true ><<<gproj, 128, 0, stream>>>(pre_v, Wv, bv, Vt16);

    attn_kernel<<<gattn, 128, 0, stream>>>(Q16, K16, Vt16, mask, A16);

    proj_gemm<true, false, true, false><<<gproj, 128, 0, stream>>>(A16, Wo, bo, d_out);
  }
}
